// SAGPoolReadout_4045859193612
// MI455X (gfx1250) — matrix operand order repaired and checked by reading, not compiled, not run on hardware
//
#include <hip/hip_runtime.h>
#include <hip/hip_bf16.h>

// ---------------------------------------------------------------------------
// SAGPool readout for MI455X (gfx1250, wave32, WMMA)
// ---------------------------------------------------------------------------

typedef __attribute__((ext_vector_type(16))) __bf16 v16bf;
typedef __attribute__((ext_vector_type(8)))  float  v8f;

#define D     128
#define NPG   1024
#define KEEP  512
#define NCLS  10

__device__ __forceinline__ float rdeg(unsigned c) {
    // clip(deg,1)^-0.5
    return rsqrtf((float)(c > 1u ? c : 1u));
}

// --- 1a: Wc (row-major f32) -> WcT (col-major bf16) -------------------------
__global__ void k_prep_wct(const float* __restrict__ Wc, __bf16* __restrict__ WcT) {
    int t = blockIdx.x * blockDim.x + threadIdx.x;
    if (t < D * D) {
        int k = t / D, n = t % D;
        WcT[n * D + k] = (__bf16)Wc[k * D + n];
    }
}

// --- 1b: degree counts ------------------------------------------------------
__global__ void k_degrees(const int* __restrict__ src, const int* __restrict__ dst,
                          unsigned* __restrict__ dout, unsigned* __restrict__ din, int E) {
    int e = blockIdx.x * blockDim.x + threadIdx.x;
    if (e < E) {
        __hip_atomic_fetch_add(&dout[src[e]], 1u, __ATOMIC_RELAXED, __HIP_MEMORY_SCOPE_AGENT);
        __hip_atomic_fetch_add(&din[dst[e]],  1u, __ATOMIC_RELAXED, __HIP_MEMORY_SCOPE_AGENT);
    }
}

// --- 2: xs = bf16( feature * deg_out^-1/2 ) --------------------------------
__global__ void k_scale_cast(const float* __restrict__ x, const unsigned* __restrict__ dout,
                             __bf16* __restrict__ xs, int n4) {
    int t = blockIdx.x * blockDim.x + threadIdx.x;
    if (t >= n4) return;
    int node = (t * 4) / D;
    float s = rdeg(dout[node]);
    float4 v = ((const float4*)x)[t];
    union { __bf16 b[4]; ushort4 u; } p;
    p.b[0] = (__bf16)(v.x * s);
    p.b[1] = (__bf16)(v.y * s);
    p.b[2] = (__bf16)(v.z * s);
    p.b[3] = (__bf16)(v.w * s);
    ((ushort4*)xs)[t] = p.u;
}

// --- 3: hlin = xs @ Wc via v_wmma_f32_16x16x32_bf16 -------------------------
// block = 256 threads = 8 waves; block computes 16 rows x 128 cols.
// wave w owns N-tile w (16 cols); 4 K-steps of 32.
__global__ void __launch_bounds__(256) k_gemm_xwc(const __bf16* __restrict__ xs,
                                                  const __bf16* __restrict__ WcT,
                                                  float* __restrict__ hlin) {
    __shared__ __align__(16) __bf16 As[16 * D];   // 4 KB A tile
    const int tid = threadIdx.x;
    const int m0  = blockIdx.x * 16;

    // cooperative A-tile load: 256 threads x 16B = 16 rows x 256B
    ((uint4*)As)[tid] = ((const uint4*)(xs + (size_t)m0 * D))[tid];
    __syncthreads();

    const int wave = tid >> 5;        // N tile
    const int lane = tid & 31;
    const int row  = lane & 15;
    const int hi   = lane >> 4;       // 0: lanes 0-15, 1: lanes 16-31

    v8f c = {};
#pragma unroll
    for (int kc = 0; kc < D; kc += 32) {
        // A frag (ISA 16-bit 16x32 layout):
        //   lanes 0-15 : K = kc+0..7,  kc+16..23   (row = lane)
        //   lanes 16-31: K = kc+8..15, kc+24..31   (row = lane-16)
        union { uint4 u[2]; v16bf v; } fa;
        const int asel = hi * 8;
        fa.u[0] = *(const uint4*)(&As[row * D + kc + asel]);
        fa.u[1] = *(const uint4*)(&As[row * D + kc + 16 + asel]);

        // B frag: lane n<16 -> col = nt*16+n, K = kc+0..7, kc+16..23;
        //         lanes 16-31 -> same cols, K = kc+8..15, kc+24..31.
        union { uint4 u[2]; v16bf v; } fb;
        const int col  = wave * 16 + row;
        const int koff = hi * 8;
        const uint4* bp = (const uint4*)(WcT + (size_t)col * D + kc + koff);
        fb.u[0] = bp[0];
        fb.u[1] = bp[2];

        c = __builtin_amdgcn_wmma_f32_16x16x32_bf16(
                false, fa.v, false, fb.v, (short)0, c, false, false);
    }

    // D layout: vgpr r -> (M = r + hi*8, N = lane&15)
    float* out = hlin + (size_t)m0 * D + wave * 16;
#pragma unroll
    for (int r = 0; r < 8; ++r) {
        out[(size_t)(r + hi * 8) * D + row] = c[r];
    }
}

// --- 4: m[dst] += hlin[src]  (1 wave per edge, lanes cover 128 floats) ------
__global__ void k_aggregate(const int* __restrict__ src, const int* __restrict__ dst,
                            const float* __restrict__ hlin, float* __restrict__ m,
                            long long T) {
    long long t = (long long)blockIdx.x * blockDim.x + threadIdx.x;
    if (t >= T) return;
    int e = (int)(t >> 5);
    int c = ((int)t & 31) * 4;
    int s = src[e], d = dst[e];
    float4 v = *(const float4*)(hlin + (size_t)s * D + c);
    float* mp = m + (size_t)d * D + c;
    __hip_atomic_fetch_add(mp + 0, v.x, __ATOMIC_RELAXED, __HIP_MEMORY_SCOPE_AGENT);
    __hip_atomic_fetch_add(mp + 1, v.y, __ATOMIC_RELAXED, __HIP_MEMORY_SCOPE_AGENT);
    __hip_atomic_fetch_add(mp + 2, v.z, __ATOMIC_RELAXED, __HIP_MEMORY_SCOPE_AGENT);
    __hip_atomic_fetch_add(mp + 3, v.w, __ATOMIC_RELAXED, __HIP_MEMORY_SCOPE_AGENT);
}

// --- 5: h = feature + relu(m * deg_in^-1/2 + bc)   (in place over m) --------
__global__ void k_finalize_h(const float* __restrict__ x, const unsigned* __restrict__ din,
                             const float* __restrict__ bc, float* __restrict__ h, int n4) {
    int t = blockIdx.x * blockDim.x + threadIdx.x;
    if (t >= n4) return;
    int elem = t * 4;
    int node = elem / D;
    int j    = elem % D;
    float s  = rdeg(din[node]);
    float4 mv = ((const float4*)h)[t];
    float4 xv = ((const float4*)x)[t];
    float4 bv = *(const float4*)(bc + j);
    float4 o;
    o.x = xv.x + fmaxf(mv.x * s + bv.x, 0.f);
    o.y = xv.y + fmaxf(mv.y * s + bv.y, 0.f);
    o.z = xv.z + fmaxf(mv.z * s + bv.z, 0.f);
    o.w = xv.w + fmaxf(mv.w * s + bv.w, 0.f);
    ((float4*)h)[t] = o;
}

// --- 6a: slin[n] = dot(h[n], Ws) * deg_out^-1/2  (1 wave32 per node) --------
__global__ void k_score_lin(const float* __restrict__ h, const float* __restrict__ Ws,
                            const unsigned* __restrict__ dout, float* __restrict__ slin,
                            int n_nodes) {
    int node = (blockIdx.x * blockDim.x + threadIdx.x) >> 5;
    int lane = threadIdx.x & 31;
    if (node >= n_nodes) return;
    float4 hv = *(const float4*)(h + (size_t)node * D + lane * 4);
    float4 wv = *(const float4*)(Ws + lane * 4);
    float v = hv.x * wv.x + hv.y * wv.y + hv.z * wv.z + hv.w * wv.w;
#pragma unroll
    for (int off = 16; off > 0; off >>= 1) v += __shfl_xor(v, off, 32);
    if (lane == 0) slin[node] = v * rdeg(dout[node]);
}

// --- 6b: sm[dst] += slin[src] ----------------------------------------------
__global__ void k_score_scatter(const int* __restrict__ src, const int* __restrict__ dst,
                                const float* __restrict__ slin, float* __restrict__ sm, int E) {
    int e = blockIdx.x * blockDim.x + threadIdx.x;
    if (e < E)
        __hip_atomic_fetch_add(&sm[dst[e]], slin[src[e]],
                               __ATOMIC_RELAXED, __HIP_MEMORY_SCOPE_AGENT);
}

// --- 7: per-graph exact top-512 (bitonic sort of 1024 in LDS) ---------------
__global__ void __launch_bounds__(1024) k_topk(const float* __restrict__ sm,
                                               const unsigned* __restrict__ din,
                                               const float* __restrict__ bs,
                                               float* __restrict__ svals,
                                               int* __restrict__ sidx) {
    __shared__ float s[NPG];
    __shared__ int   id[NPG];
    const int g = blockIdx.x, t = threadIdx.x;
    const int node = g * NPG + t;
    s[t]  = sm[node] * rdeg(din[node]) + bs[0];
    id[t] = t;
    __syncthreads();
    for (int k = 2; k <= NPG; k <<= 1) {
        for (int j = k >> 1; j > 0; j >>= 1) {
            int ixj = t ^ j;
            if (ixj > t) {
                bool dir = ((t & k) == 0);            // final pass: descending
                float si = s[t], sj = s[ixj];
                int   ii = id[t], ij = id[ixj];
                bool iBefore = (si > sj) || (si == sj && ii < ij);
                if (iBefore != dir) { s[t] = sj; s[ixj] = si; id[t] = ij; id[ixj] = ii; }
            }
            __syncthreads();
        }
    }
    if (t < KEEP) {
        svals[g * KEEP + t] = s[t];
        sidx [g * KEEP + t] = id[t];
    }
}

// --- 8: gated mean/max readout: hg[g] = [mean || max] of h[idx]*tanh(val) ---
__global__ void __launch_bounds__(128) k_readout(const float* __restrict__ h,
                                                 const float* __restrict__ svals,
                                                 const int* __restrict__ sidx,
                                                 float* __restrict__ hg) {
    __shared__ float tg[KEEP];
    __shared__ int   li[KEEP];
    const int g = blockIdx.x, t = threadIdx.x;
    for (int i = t; i < KEEP; i += 128) {
        tg[i] = tanhf(svals[g * KEEP + i]);
        li[i] = sidx[g * KEEP + i];
    }
    __syncthreads();
    float sum = 0.f, mx = -3.402823466e38f;
    for (int i = 0; i < KEEP; ++i) {
        float x = h[((size_t)g * NPG + li[i]) * D + t] * tg[i];
        sum += x;
        mx = fmaxf(mx, x);
    }
    hg[g * (2 * D) + t]     = sum * (1.0f / KEEP);
    hg[g * (2 * D) + D + t] = mx;
}

// --- 9: MLP readout 256 -> 128 -> 64 -> 10 ----------------------------------
__global__ void __launch_bounds__(256) k_mlp(const float* __restrict__ hg,
                                             const float* __restrict__ W0, const float* __restrict__ b0,
                                             const float* __restrict__ W1, const float* __restrict__ b1,
                                             const float* __restrict__ W2, const float* __restrict__ b2,
                                             float* __restrict__ out) {
    __shared__ float in[2 * D];
    __shared__ float h0[D];
    __shared__ float h1[D / 2];
    const int g = blockIdx.x, t = threadIdx.x;
    in[t] = hg[g * 2 * D + t];
    __syncthreads();
    if (t < D) {
        float acc = b0[t];
        for (int k = 0; k < 2 * D; ++k) acc += in[k] * W0[k * D + t];
        h0[t] = fmaxf(acc, 0.f);
    }
    __syncthreads();
    if (t < D / 2) {
        float acc = b1[t];
        for (int k = 0; k < D; ++k) acc += h0[k] * W1[k * (D / 2) + t];
        h1[t] = fmaxf(acc, 0.f);
    }
    __syncthreads();
    if (t < NCLS) {
        float acc = b2[t];
        for (int k = 0; k < D / 2; ++k) acc += h1[k] * W2[k * NCLS + t];
        out[g * NCLS + t] = acc;
    }
}

// ---------------------------------------------------------------------------
extern "C" void kernel_launch(void* const* d_in, const int* in_sizes, int n_in,
                              void* d_out, int out_size, void* d_ws, size_t ws_size,
                              hipStream_t stream) {
    const float* feature = (const float*)d_in[0];
    const int*   src     = (const int*)  d_in[1];
    const int*   dst     = (const int*)  d_in[2];
    const float* Wc      = (const float*)d_in[3];
    const float* bc      = (const float*)d_in[4];
    const float* Ws      = (const float*)d_in[5];
    const float* bs      = (const float*)d_in[6];
    const float* W0      = (const float*)d_in[7];
    const float* b0      = (const float*)d_in[8];
    const float* W1      = (const float*)d_in[9];
    const float* b1      = (const float*)d_in[10];
    const float* W2      = (const float*)d_in[11];
    const float* b2      = (const float*)d_in[12];
    // d_in[13] = n_per_graph (device scalar); NPG is compile-time 1024.

    const int n_nodes = in_sizes[0] / D;
    const int E       = in_sizes[1];
    const int B       = n_nodes / NPG;

    // ---- carve workspace ----
    char*  ws = (char*)d_ws;
    size_t o  = 0;
    auto alloc = [&](size_t bytes) -> char* {
        char* p = ws + o;
        o = (o + bytes + 255) & ~(size_t)255;
        return p;
    };
    __bf16*   WcT   = (__bf16*)  alloc((size_t)D * D * sizeof(__bf16));
    unsigned* dout  = (unsigned*)alloc((size_t)n_nodes * 4);
    unsigned* din   = (unsigned*)alloc((size_t)n_nodes * 4);
    float*    sm    = (float*)   alloc((size_t)n_nodes * 4);
    float*    svals = (float*)   alloc((size_t)B * KEEP * 4);
    int*      sidx  = (int*)     alloc((size_t)B * KEEP * 4);
    float*    hg    = (float*)   alloc((size_t)B * 2 * D * 4);
    __bf16*   xs    = (__bf16*)  alloc((size_t)n_nodes * D * sizeof(__bf16));
    float*    hlin  = (float*)   alloc((size_t)n_nodes * D * 4);
    float*    m     = (float*)   alloc((size_t)n_nodes * D * 4);  // becomes h in place
    (void)ws_size; (void)n_in; (void)out_size;

    // ---- zero accumulators (capture-safe async memsets) ----
    hipMemsetAsync(dout, 0, (size_t)n_nodes * 4, stream);
    hipMemsetAsync(din,  0, (size_t)n_nodes * 4, stream);
    hipMemsetAsync(sm,   0, (size_t)n_nodes * 4, stream);
    hipMemsetAsync(m,    0, (size_t)n_nodes * D * 4, stream);

    const int n4 = n_nodes * D / 4;

    k_prep_wct   <<<(D * D + 255) / 256, 256, 0, stream>>>(Wc, WcT);
    k_degrees    <<<(E + 255) / 256, 256, 0, stream>>>(src, dst, dout, din, E);
    k_scale_cast <<<(n4 + 255) / 256, 256, 0, stream>>>(feature, dout, xs, n4);
    k_gemm_xwc   <<<n_nodes / 16, 256, 0, stream>>>(xs, WcT, hlin);

    long long T = (long long)E * 32;
    k_aggregate  <<<(int)((T + 255) / 256), 256, 0, stream>>>(src, dst, hlin, m, T);
    k_finalize_h <<<(n4 + 255) / 256, 256, 0, stream>>>(feature, din, bc, m, n4);

    float* slin = hlin;  // hlin free after aggregation
    k_score_lin    <<<n_nodes / 8, 256, 0, stream>>>(m, Ws, dout, slin, n_nodes);
    k_score_scatter<<<(E + 255) / 256, 256, 0, stream>>>(src, dst, slin, sm, E);
    k_topk         <<<B, 1024, 0, stream>>>(sm, din, bs, svals, sidx);
    k_readout      <<<B, 128, 0, stream>>>(m, svals, sidx, hg);
    k_mlp          <<<B, 256, 0, stream>>>(hg, W0, b0, W1, b1, W2, b2, (float*)d_out);
}
